// MultiHeadAttention_27736898798340
// MI455X (gfx1250) — compile-verified, matrix operand order checked by reading, not run on hardware
//
#include <hip/hip_runtime.h>

typedef __attribute__((ext_vector_type(16))) _Float16 v16h;
typedef __attribute__((ext_vector_type(8)))  _Float16 h8;
typedef __attribute__((ext_vector_type(4)))  _Float16 h4;
typedef __attribute__((ext_vector_type(8)))  float    v8f;

#if __has_builtin(__builtin_amdgcn_tensor_load_to_lds) && __has_builtin(__builtin_amdgcn_s_wait_tensorcnt)
#define USE_TDM 1
typedef __attribute__((ext_vector_type(4))) unsigned int u32x4;
typedef __attribute__((ext_vector_type(4))) int          i32x4;
typedef __attribute__((ext_vector_type(8))) int          i32x8;
#endif

#define B_    8
#define C_    256
#define P_    8192
#define H_    8
#define F_    64
#define S_    64
#define SS_   4096
#define O_    536      // H*(F+3)
#define OP_   544      // padded to 34 * 16
#define OKEY_ 24       // H*3
#define BH_   64
#define NBP_  65536    // B_*P_
#define EPS_  1e-5f

__device__ __forceinline__ v16h cat16(h8 lo, h8 hi) {
  return __builtin_shufflevector(lo, hi, 0, 1, 2, 3, 4, 5, 6, 7,
                                 8, 9, 10, 11, 12, 13, 14, 15);
}

// ---------------- precision prep ---------------------------------------------------
__global__ void k_cvt_x(const float* __restrict__ X, _Float16* __restrict__ Xh) {
  size_t i = ((size_t)blockIdx.x * 256 + threadIdx.x) * 4;
  float4 v = *(const float4*)(X + i);
  h4 o; o[0] = (_Float16)v.x; o[1] = (_Float16)v.y;
        o[2] = (_Float16)v.z; o[3] = (_Float16)v.w;
  *(h4*)(Xh + i) = o;
}

__global__ void k_cvt_w(const float* __restrict__ W, _Float16* __restrict__ Wh) {
  int t = blockIdx.x * 256 + threadIdx.x;   // OP_*C_
  int o = t >> 8, c = t & 255;
  Wh[t] = (o < O_) ? (_Float16)W[o * C_ + c] : (_Float16)0.f;
}

// conv_w (H*F, F, 3, 3) -> CWt[h][kpos][f][ci] as f16 (contiguous in ci)
__global__ void k_prepw(const float* __restrict__ CW, _Float16* __restrict__ CWt) {
  int t = blockIdx.x * 256 + threadIdx.x;   // 8*9*64*64
  int ci = t & 63, f = (t >> 6) & 63;
  int rest = t >> 12, kpos = rest % 9, h = rest / 9;
  CWt[t] = (_Float16)CW[(((size_t)(h * F_ + f)) * F_ + ci) * 9 + kpos];
}

// ---------------- kv = W_kv @ input : WMMA, 16x64 strip per wave --------------------
__global__ void k_gemm_kv(const _Float16* __restrict__ Xh, const _Float16* __restrict__ Wh,
                          float* __restrict__ KV) {
  int wid  = (blockIdx.x * blockDim.x + threadIdx.x) >> 5;
  int lane = threadIdx.x & 31;
  int ng    = wid & 127;
  int rem   = wid >> 7;
  int mtile = rem % 34;
  int b     = rem / 34;
  int p0 = ng << 6, o0 = mtile << 4;
  int m   = lane & 15;
  int khi = (lane >> 4) << 3;
  int o = o0 + m;
  const _Float16* Xb = Xh + (size_t)b * C_ * P_;
  const _Float16* Wr = Wh + (size_t)o * C_;
  v8f acc[4] = {};
  for (int kc = 0; kc < C_; kc += 32) {
    v16h a = cat16(*(const h8*)(Wr + kc + khi),
                   *(const h8*)(Wr + kc + 16 + khi));
    if (kc + 32 < C_)
      __builtin_prefetch(Xb + (size_t)(kc + 32 + khi) * P_ + p0 + m * 2, 0, 1);
#pragma unroll
    for (int t = 0; t < 4; ++t) {
      int pl = p0 + (t << 4) + m;
      v16h bb;
#pragma unroll
      for (int j = 0; j < 16; ++j) {
        int k = (j < 8) ? (khi + j) : (16 + khi + (j - 8));
        bb[j] = Xb[(size_t)(kc + k) * P_ + pl];
      }
      acc[t] = __builtin_amdgcn_wmma_f32_16x16x32_f16(false, a, false, bb, (short)0,
                                                      acc[t], false, false);
    }
  }
#pragma unroll
  for (int t = 0; t < 4; ++t)
#pragma unroll
    for (int r = 0; r < 8; ++r) {
      int oo = o0 + r + ((lane >> 4) << 3);
      KV[((size_t)b * OP_ + oo) * P_ + p0 + (t << 4) + m] = acc[t][r];
    }
}

// ---------------- per-channel sum / sumsq of kv over (B,P) ------------------------
__global__ void k_kvstats(const float* __restrict__ KV, float* __restrict__ ksum,
                          float* __restrict__ ksq) {
  int o = blockIdx.x;
  float s = 0.f, q = 0.f;
  for (int i = threadIdx.x; i < NBP_; i += 256) {
    int b = i >> 13, p = i & (P_ - 1);
    float v = KV[((size_t)b * OP_ + o) * P_ + p];
    s += v; q += v * v;
  }
  __shared__ float ss[256], sq[256];
  ss[threadIdx.x] = s; sq[threadIdx.x] = q;
  __syncthreads();
  for (int st = 128; st > 0; st >>= 1) {
    if (threadIdx.x < st) { ss[threadIdx.x] += ss[threadIdx.x + st];
                            sq[threadIdx.x] += sq[threadIdx.x + st]; }
    __syncthreads();
  }
  if (threadIdx.x == 0) { ksum[o] = ss[0]; ksq[o] = sq[0]; }
}

__global__ void k_kvfinal(const float* ksum, const float* ksq, const float* vg,
                          const float* vb, const float* kg, const float* kb,
                          float* scale, float* shift) {
  int o = blockIdx.x * blockDim.x + threadIdx.x;
  if (o >= O_) return;
  float mean = ksum[o] * (1.f / NBP_);
  float var  = ksq[o] * (1.f / NBP_) - mean * mean;
  float inv  = rsqrtf(var + EPS_);
  float g, bt;
  if (o < OKEY_) { g = kg[o]; bt = kb[o]; }
  else           { g = vg[o - OKEY_]; bt = vb[o - OKEY_]; }
  float sc = inv * g;
  scale[o] = sc;
  shift[o] = bt - mean * sc;
}

__global__ void k_zero(float* g, size_t n) {
  size_t i = (size_t)blockIdx.x * blockDim.x + threadIdx.x;
  if (i < n) g[i] = 0.f;
}

// ---------------- lattice coords, bilinear weights & indices ----------------------
__global__ void k_point(const float* __restrict__ KV, const float* __restrict__ orig,
                        const float* __restrict__ scale, const float* __restrict__ shift,
                        const float* __restrict__ M, float* __restrict__ wbuf,
                        int* __restrict__ ibuf) {
  int t  = blockIdx.x * blockDim.x + threadIdx.x;
  int p  = t & (P_ - 1);
  int bh = t >> 13;
  int b = bh >> 3, h = bh & 7;
  float pt[3];
#pragma unroll
  for (int d = 0; d < 3; ++d) {
    int ch = h * 3 + d;
    float off = KV[((size_t)b * OP_ + ch) * P_ + p] * scale[ch] + shift[ch];
    pt[d] = orig[((size_t)b * 3 + d) * P_ + p] + off;
  }
  int basei[2]; float frac[2];
#pragma unroll
  for (int dd = 0; dd < 2; ++dd) {
    const float* Mr = M + h * 6 + dd * 3;
    float key = Mr[0] * pt[0] + Mr[1] * pt[1] + Mr[2] * pt[2];
    float pos = (tanhf(key) + 1.f) * (0.5f * (S_ - 1));
    float bse = fminf(fmaxf(floorf(pos), 0.f), (float)(S_ - 2));
    basei[dd] = (int)bse;
    frac[dd]  = pos - bse;
  }
  int baseflat = basei[0] * S_ + basei[1];
  float fx = frac[0], fy = frac[1];
  size_t wb = (size_t)bh * 4 * P_ + p;
  wbuf[wb]          = (1.f - fx) * (1.f - fy);
  wbuf[wb + P_]     = (1.f - fx) * fy;
  wbuf[wb + 2 * P_] = fx * (1.f - fy);
  wbuf[wb + 3 * P_] = fx * fy;
  ibuf[wb]          = baseflat;
  ibuf[wb + P_]     = baseflat + 1;
  ibuf[wb + 2 * P_] = baseflat + S_;
  ibuf[wb + 3 * P_] = baseflat + S_ + 1;
}

// ---------------- scatter splat: gridT[bh][cell][f] += w * value ------------------
__global__ void k_splat(const float* __restrict__ KV, const float* __restrict__ scale,
                        const float* __restrict__ shift, const float* __restrict__ wbuf,
                        const int* __restrict__ ibuf, float* __restrict__ gridT) {
  int t  = blockIdx.x * blockDim.x + threadIdx.x;
  int p  = t & (P_ - 1);
  int bh = t >> 13;
  int b = bh >> 3, h = bh & 7;
  size_t wb = (size_t)bh * 4 * P_ + p;
  float w0 = wbuf[wb], w1 = wbuf[wb + P_], w2 = wbuf[wb + 2 * P_], w3 = wbuf[wb + 3 * P_];
  int   i0 = ibuf[wb], i1 = ibuf[wb + P_], i2 = ibuf[wb + 2 * P_], i3 = ibuf[wb + 3 * P_];
  float* g = gridT + (size_t)bh * SS_ * F_;
  for (int f = 0; f < F_; ++f) {
    int ch = OKEY_ + h * F_ + f;
    float val = KV[((size_t)b * OP_ + ch) * P_ + p] * scale[ch] + shift[ch];
    atomicAdd(&g[(size_t)i0 * F_ + f], w0 * val);
    atomicAdd(&g[(size_t)i1 * F_ + f], w1 * val);
    atomicAdd(&g[(size_t)i2 * F_ + f], w2 * val);
    atomicAdd(&g[(size_t)i3 * F_ + f], w3 * val);
  }
}

// ---------------- grouped 3x3 conv: TDM row staging into LDS + WMMA ---------------
// one block per (bh, y): LDS holds rows y-1,y,y+1 as [3][66 cells][64 ci] fp32.
// wave w: ntile = w&3 (16 pixels), m-tile pair = w>>2 (two 16-row f tiles).
__global__ void k_conv(const float* __restrict__ gridT, const _Float16* __restrict__ CWt,
                       const float* __restrict__ CB, float* __restrict__ convout) {
  __shared__ __align__(16) float tile[3 * 66 * 64];
  int tid  = threadIdx.x;
  int lane = tid & 31;
  int wv   = tid >> 5;
  int bh = blockIdx.x >> 6;
  int y  = blockIdx.x & 63;
  int h = bh & 7;
  const float* Grow = gridT + (size_t)bh * SS_ * F_;

#ifdef USE_TDM
  if (wv == 0) {
    unsigned ldsbase = (unsigned)(size_t)&tile[0];
    for (int row = 0; row < 3; ++row) {
      long long cell0 = (long long)(y + row - 1) * S_ - 1;    // halo starts at x=-1
      unsigned long long ga = (unsigned long long)(const void*)(Grow + cell0 * F_);
      u32x4 g0;
      g0[0] = 1u;                                  // count = 1 (valid 2D descriptor)
      g0[1] = ldsbase + (unsigned)(row * 66 * 64 * 4);
      g0[2] = (unsigned)ga;
      g0[3] = (unsigned)(ga >> 32) | (2u << 30);   // global_addr[56:32] | type=2
      i32x8 g1;
      g1[0] = 0x00020000;                          // data_size = 4 bytes
      g1[1] = (64 << 16);                          // tensor_dim0 = 64 (ci)
      g1[2] = 0;                                   // tensor_dim1[15:0] = 0
      g1[3] = 0x4000 | (64 << 16);                 // tensor_dim1 = 1<<30; tile_dim0 = 64
      g1[4] = 66;                                  // tile_dim1 = 66 cells
      g1[5] = 64;                                  // tensor_dim0_stride = 64
      g1[6] = 0; g1[7] = 0;
      i32x4 gz = {0, 0, 0, 0};
#if defined(__clang_major__) && (__clang_major__ >= 23)
      i32x8 gz8 = {0, 0, 0, 0, 0, 0, 0, 0};
      __builtin_amdgcn_tensor_load_to_lds(g0, g1, gz, gz, gz8, 0);
#else
      __builtin_amdgcn_tensor_load_to_lds(g0, g1, gz, gz, 0);
#endif
    }
    __builtin_amdgcn_s_wait_tensorcnt(0);
  }
  __syncthreads();
#else
  for (int i = tid; i < 3 * 66 * 16; i += 256) {
    int row = i / (66 * 16);
    int rem = i % (66 * 16);
    int slot = rem >> 4, quad = rem & 15;
    long long cell = (long long)(y + row - 1) * S_ + (slot - 1);
    float4 v = *(const float4*)(Grow + cell * F_ + quad * 4);
    *(float4*)(&tile[(row * 66 + slot) * 64 + quad * 4]) = v;
  }
  __syncthreads();
#endif

  int ntile = wv & 3;
  int mpair = wv >> 2;
  int x0  = ntile << 4;
  int n   = lane & 15;
  int khi = (lane >> 4) << 3;
  int frow0 = (mpair * 32) + n;          // m-tile pair: rows frow0, frow0+16
  const _Float16* Wt = CWt + (size_t)h * 9 * F_ * F_;
  v8f acc0 = {}, acc1 = {};
#pragma unroll
  for (int kpos = 0; kpos < 9; ++kpos) {
    int ky = kpos / 3, kx = kpos % 3;
    int yy = y + ky - 1;
    int xx = x0 + n + kx - 1;
    bool valid = (yy >= 0) & (yy < S_) & (xx >= 0) & (xx < S_);
    const float* src = &tile[((ky * 66) + x0 + n + kx) * 64];
    const _Float16* wr0 = Wt + ((size_t)kpos * F_ + frow0) * F_;
#pragma unroll
    for (int half = 0; half < 2; ++half) {
      int ci0 = half << 5;
      float4 f0 = {0.f, 0.f, 0.f, 0.f}, f1 = f0, f2 = f0, f3 = f0;
      if (valid) {
        f0 = *(const float4*)(src + ci0 + khi);
        f1 = *(const float4*)(src + ci0 + khi + 4);
        f2 = *(const float4*)(src + ci0 + 16 + khi);
        f3 = *(const float4*)(src + ci0 + 16 + khi + 4);
      }
      v16h bb;
      bb[0]  = (_Float16)f0.x; bb[1]  = (_Float16)f0.y;
      bb[2]  = (_Float16)f0.z; bb[3]  = (_Float16)f0.w;
      bb[4]  = (_Float16)f1.x; bb[5]  = (_Float16)f1.y;
      bb[6]  = (_Float16)f1.z; bb[7]  = (_Float16)f1.w;
      bb[8]  = (_Float16)f2.x; bb[9]  = (_Float16)f2.y;
      bb[10] = (_Float16)f2.z; bb[11] = (_Float16)f2.w;
      bb[12] = (_Float16)f3.x; bb[13] = (_Float16)f3.y;
      bb[14] = (_Float16)f3.z; bb[15] = (_Float16)f3.w;
      v16h a0 = cat16(*(const h8*)(wr0 + ci0 + khi),
                      *(const h8*)(wr0 + ci0 + 16 + khi));
      v16h a1 = cat16(*(const h8*)(wr0 + 16 * F_ + ci0 + khi),
                      *(const h8*)(wr0 + 16 * F_ + ci0 + 16 + khi));
      acc0 = __builtin_amdgcn_wmma_f32_16x16x32_f16(false, a0, false, bb, (short)0, acc0,
                                                    false, false);
      acc1 = __builtin_amdgcn_wmma_f32_16x16x32_f16(false, a1, false, bb, (short)0, acc1,
                                                    false, false);
    }
  }
  float* out = convout + (size_t)bh * SS_ * F_ + (size_t)(y * S_ + x0 + n) * F_;
  int hi8 = (lane >> 4) << 3;
#pragma unroll
  for (int r = 0; r < 8; ++r) {
    int f0i = mpair * 32 + r + hi8;
    out[f0i]      = acc0[r] + CB[h * F_ + f0i];
    out[f0i + 16] = acc1[r] + CB[h * F_ + f0i + 16];
  }
}

// ---------------- gather + BN statistics ------------------------------------------
__global__ void k_gstats(const float* __restrict__ convout, const float* __restrict__ wbuf,
                         const int* __restrict__ ibuf, float* __restrict__ asum,
                         float* __restrict__ asq) {
  int bh    = blockIdx.x & 63;
  int chunk = blockIdx.x >> 6;
  int f  = threadIdx.x;
  int ty = threadIdx.y;
  int h = bh & 7;
  const float* CO = convout + (size_t)bh * SS_ * F_;
  const float* Wp = wbuf + (size_t)bh * 4 * P_;
  const int*   Ip = ibuf + (size_t)bh * 4 * P_;
  float s = 0.f, q = 0.f;
  for (int i = 0; i < 256; ++i) {
    int p = chunk * 1024 + ty + 4 * i;
    float w0 = Wp[p], w1 = Wp[p + P_], w2 = Wp[p + 2 * P_], w3 = Wp[p + 3 * P_];
    int   i0 = Ip[p], i1 = Ip[p + P_], i2 = Ip[p + 2 * P_], i3 = Ip[p + 3 * P_];
    float v = w0 * CO[(size_t)i0 * F_ + f] + w1 * CO[(size_t)i1 * F_ + f] +
              w2 * CO[(size_t)i2 * F_ + f] + w3 * CO[(size_t)i3 * F_ + f];
    s += v; q += v * v;
  }
  __shared__ float ss[4][64], sq2[4][64];
  ss[ty][f] = s; sq2[ty][f] = q;
  __syncthreads();
  if (ty == 0) {
    atomicAdd(&asum[h * F_ + f], ss[0][f] + ss[1][f] + ss[2][f] + ss[3][f]);
    atomicAdd(&asq[h * F_ + f], sq2[0][f] + sq2[1][f] + sq2[2][f] + sq2[3][f]);
  }
}

__global__ void k_afinal(const float* asum, const float* asq, const float* ag,
                         const float* ab, float* scale, float* shift) {
  int c = blockIdx.x * blockDim.x + threadIdx.x;
  if (c >= H_ * F_) return;
  float mean = asum[c] * (1.f / NBP_);
  float var  = asq[c] * (1.f / NBP_) - mean * mean;
  float inv  = rsqrtf(var + EPS_);
  float sc = inv * ag[c];
  scale[c] = sc;
  shift[c] = ab[c] - mean * sc;
}

// ---------------- gather again, BN + ReLU, write output ---------------------------
__global__ void k_output(const float* __restrict__ convout, const float* __restrict__ wbuf,
                         const int* __restrict__ ibuf, const float* __restrict__ ascale,
                         const float* __restrict__ ashift, float* __restrict__ out) {
  int t  = blockIdx.x * blockDim.x + threadIdx.x;
  int p  = t & (P_ - 1);
  int bh = t >> 13;
  int b = bh >> 3, h = bh & 7;
  const float* CO = convout + (size_t)bh * SS_ * F_;
  size_t wb = (size_t)bh * 4 * P_ + p;
  float w0 = wbuf[wb], w1 = wbuf[wb + P_], w2 = wbuf[wb + 2 * P_], w3 = wbuf[wb + 3 * P_];
  int   i0 = ibuf[wb], i1 = ibuf[wb + P_], i2 = ibuf[wb + 2 * P_], i3 = ibuf[wb + 3 * P_];
  for (int f = 0; f < F_; ++f) {
    float v = w0 * CO[(size_t)i0 * F_ + f] + w1 * CO[(size_t)i1 * F_ + f] +
              w2 * CO[(size_t)i2 * F_ + f] + w3 * CO[(size_t)i3 * F_ + f];
    int ch = h * F_ + f;
    v = v * ascale[ch] + ashift[ch];
    out[((size_t)b * (H_ * F_) + ch) * P_ + p] = fmaxf(v, 0.f);
  }
}

extern "C" void kernel_launch(void* const* d_in, const int* in_sizes, int n_in,
                              void* d_out, int out_size, void* d_ws, size_t ws_size,
                              hipStream_t stream) {
  const float* input  = (const float*)d_in[0];
  const float* orig   = (const float*)d_in[1];
  const float* W_kv   = (const float*)d_in[2];
  const float* vgam   = (const float*)d_in[3];
  const float* vbet   = (const float*)d_in[4];
  const float* kgam   = (const float*)d_in[5];
  const float* kbet   = (const float*)d_in[6];
  const float* Mtr    = (const float*)d_in[7];
  const float* convw  = (const float*)d_in[8];
  const float* convb  = (const float*)d_in[9];
  const float* agam   = (const float*)d_in[10];
  const float* abet   = (const float*)d_in[11];
  float* out = (float*)d_out;

  const size_t nKV   = (size_t)B_ * OP_ * P_;
  const size_t nW    = (size_t)BH_ * 4 * P_;
  const size_t nGrid = (size_t)BH_ * SS_ * F_;
  const size_t nX    = (size_t)B_ * C_ * P_;
  char* w = (char*)d_ws;
  float*    KV    = (float*)w;      w += nKV * 4;
  float*    wbuf  = (float*)w;      w += nW * 4;
  int*      ibuf  = (int*)w;        w += nW * 4;
  float*    gridT = (float*)w;      w += nGrid * 4;
  _Float16* Xh    = (_Float16*)w;   w += nX * 2;
  _Float16* Wh    = (_Float16*)w;   w += (size_t)OP_ * C_ * 2;
  _Float16* CWt   = (_Float16*)w;   w += (size_t)H_ * 9 * F_ * F_ * 2;
  float* ksum = (float*)w;          w += O_ * 4;
  float* ksq  = (float*)w;          w += O_ * 4;
  float* kscl = (float*)w;          w += O_ * 4;
  float* ksft = (float*)w;          w += O_ * 4;
  float* asum = (float*)w;          w += 512 * 4;
  float* asq  = (float*)w;          w += 512 * 4;
  float* ascl = (float*)w;          w += 512 * 4;
  float* asft = (float*)w;          w += 512 * 4;
  float* convout = KV;              // kv dead after splat; alias

  k_cvt_x<<<(int)(nX / 4 / 256), 256, 0, stream>>>(input, Xh);
  k_cvt_w<<<(OP_ * C_) / 256, 256, 0, stream>>>(W_kv, Wh);
  k_prepw<<<(H_ * 9 * F_ * F_) / 256, 256, 0, stream>>>(convw, CWt);

  k_zero<<<(int)((nGrid + 255) / 256), 256, 0, stream>>>(gridT, nGrid);
  k_zero<<<4, 256, 0, stream>>>(asum, 1024);

  k_gemm_kv<<<(8 * 34 * 128) / 8, 256, 0, stream>>>(Xh, Wh, KV);

  k_kvstats<<<O_, 256, 0, stream>>>(KV, ksum, ksq);
  k_kvfinal<<<3, 256, 0, stream>>>(ksum, ksq, vgam, vbet, kgam, kbet, kscl, ksft);

  k_point<<<(BH_ * P_) / 256, 256, 0, stream>>>(KV, orig, kscl, ksft, Mtr, wbuf, ibuf);
  k_splat<<<(BH_ * P_) / 256, 256, 0, stream>>>(KV, kscl, ksft, wbuf, ibuf, gridT);

  // conv: one block per (bh, y) row; TDM stages 3 input rows into LDS
  k_conv<<<BH_ * S_, 256, 0, stream>>>(gridT, CWt, convb, convout);

  k_gstats<<<BH_ * 8, dim3(64, 4), 0, stream>>>(convout, wbuf, ibuf, asum, asq);
  k_afinal<<<2, 256, 0, stream>>>(asum, asq, agam, abet, ascl, asft);

  k_output<<<(BH_ * P_) / 256, 256, 0, stream>>>(convout, wbuf, ibuf, ascl, asft, out);
}
